// SlidingWindowPatchVoxelFusion_46119358824944
// MI455X (gfx1250) — matrix operand order repaired and checked by reading, not compiled, not run on hardware
//
#include <hip/hip_runtime.h>
#include <hip/hip_bf16.h>
#include <math.h>

// ---------------------------------------------------------------------------
// Types / WMMA helpers (CDNA5 gfx1250, wave32)
// ---------------------------------------------------------------------------
typedef __attribute__((ext_vector_type(16))) _Float16 v16h;
typedef __attribute__((ext_vector_type(2)))  _Float16 h2v;
typedef __attribute__((ext_vector_type(8)))  float    v8f;

// A-matrix (16x32 f16) K index for element e, lane half = lane>>4
__device__ __forceinline__ int kmapA(int e, int half) {
    return e + half * 16;
}
// B-matrix (32x16 f16) K index for element e
__device__ __forceinline__ int kmapB(int e, int half) {
    return e + half * 16;
}

struct Frag { v16h hi; v16h lo; };

// A fragment gather from row-major f32 tile (split done here)
__device__ __forceinline__ Frag load_fragA(const float* p, int stride, int row,
                                           int kbase, int half) {
    Frag f;
#pragma unroll
    for (int e = 0; e < 16; ++e) {
        float v = p[row * stride + kbase + kmapA(e, half)];
        _Float16 h = (_Float16)v;
        f.hi[e] = h;
        f.lo[e] = (_Float16)(v - (float)h);
    }
    return f;
}

// B fragment gather from K-major f32 matrix (global weights)
__device__ __forceinline__ Frag load_fragB(const float* p, int stride, int col,
                                           int kbase, int half) {
    Frag f;
#pragma unroll
    for (int e = 0; e < 16; ++e) {
        float v = p[(size_t)(kbase + kmapB(e, half)) * stride + col];
        _Float16 h = (_Float16)v;
        f.hi[e] = h;
        f.lo[e] = (_Float16)(v - (float)h);
    }
    return f;
}

// B fragment gather from N-major (transposed) f32 tile: B[k][n] = p[n*stride+k]
__device__ __forceinline__ Frag load_fragBT(const float* p, int stride, int col,
                                            int kbase, int half) {
    Frag f;
#pragma unroll
    for (int e = 0; e < 16; ++e) {
        float v = p[col * stride + kbase + kmapB(e, half)];
        _Float16 h = (_Float16)v;
        f.hi[e] = h;
        f.lo[e] = (_Float16)(v - (float)h);
    }
    return f;
}

// B fragment gather from pre-split N-major f16 hi/lo planes: K contiguous,
// so each pair of elements is one 32-bit DS load, zero VALU conversion.
__device__ __forceinline__ Frag load_fragB16(const _Float16* ph,
                                             const _Float16* pl, int stride,
                                             int col, int kbase, int half) {
    Frag f;
#pragma unroll
    for (int p = 0; p < 8; ++p) {
        int k = kbase + 2 * p + 16 * half;
        h2v vh = *(const h2v*)(ph + col * stride + k);
        h2v vl = *(const h2v*)(pl + col * stride + k);
        f.hi[2 * p]     = vh.x;
        f.hi[2 * p + 1] = vh.y;
        f.lo[2 * p]     = vl.x;
        f.lo[2 * p + 1] = vl.y;
    }
    return f;
}

// fp32-accurate product via 3 f16 WMMAs: (ah+al)*(bh+bl) ~ ah*bh + ah*bl + al*bh
__device__ __forceinline__ v8f mma3(const Frag& a, const Frag& b, v8f c) {
    c = __builtin_amdgcn_wmma_f32_16x16x32_f16(false, a.hi, false, b.hi,
                                               (short)0, c, false, false);
    c = __builtin_amdgcn_wmma_f32_16x16x32_f16(false, a.hi, false, b.lo,
                                               (short)0, c, false, false);
    c = __builtin_amdgcn_wmma_f32_16x16x32_f16(false, a.lo, false, b.hi,
                                               (short)0, c, false, false);
    return c;
}

// ---------------------------------------------------------------------------
// CDNA5 async global->LDS copy (ASYNCcnt-tracked), 16 bytes per lane.
// ---------------------------------------------------------------------------
__device__ __forceinline__ void async_load_b128(const void* gptr,
                                                unsigned lds_off) {
    asm volatile("global_load_async_to_lds_b128 %0, %1, off"
                 :
                 : "v"(lds_off), "v"(gptr)
                 : "memory");
}
__device__ __forceinline__ void async_wait0() {
    asm volatile("s_wait_asynccnt 0" ::: "memory");
}
__device__ __forceinline__ unsigned lds_offset_of(const void* p) {
    return (unsigned)(unsigned long long)p;
}

__device__ __forceinline__ float gelu_exact(float x) {
    return 0.5f * x * (1.0f + erff(x * 0.70710678118654752f));
}

// Problem constants
#define CDIM 512
#define SDIM 512     // A(256 patch) + VCAP(256 voxel)
#define NWIN 64      // 8x8 windows
#define NVOX 2048
#define PDIM 64      // patch grid 64x64
#define NROWS (NWIN * SDIM)  // 32768

// ---------------------------------------------------------------------------
// k_zero: zero output accumulator + counts
// ---------------------------------------------------------------------------
__global__ void k_zero(float* __restrict__ out, float* __restrict__ counts) {
    int i = blockIdx.x * 256 + threadIdx.x;
    if (i < PDIM * PDIM * CDIM) out[i] = 0.0f;
    if (i < PDIM * PDIM) counts[i] = 0.0f;
}

// ---------------------------------------------------------------------------
// k_gather: per-window token build (patch window + voxel membership compaction)
// ---------------------------------------------------------------------------
__global__ void k_gather(const float* __restrict__ patch,
                         const float* __restrict__ vox,
                         const int* __restrict__ vpy,
                         const int* __restrict__ vpx,
                         float* __restrict__ tokens,
                         int* __restrict__ padm) {
    __shared__ int s_cnt[256];
    __shared__ int s_list[256];
    __shared__ int s_total;
    int w = blockIdx.x, tid = threadIdx.x;
    int y0 = (w >> 3) * 8, x0 = (w & 7) * 8;

    // membership over 2048 voxels, 8 contiguous per thread (preserves order)
    int base = tid * 8, cnt = 0;
    int loc[8];
#pragma unroll
    for (int i = 0; i < 8; ++i) {
        int vid = base + i;
        int py = vpy[vid], px = vpx[vid];
        bool m = (py >= y0) && (py < y0 + 16) && (px >= x0) && (px < x0 + 16);
        loc[i] = m ? vid : -1;
        cnt += m ? 1 : 0;
    }
    s_cnt[tid] = cnt;
    __syncthreads();
    for (int d = 1; d < 256; d <<= 1) {  // inclusive Hillis-Steele scan
        int v = s_cnt[tid];
        int add = (tid >= d) ? s_cnt[tid - d] : 0;
        __syncthreads();
        s_cnt[tid] = v + add;
        __syncthreads();
    }
    int off = s_cnt[tid] - cnt;
    if (tid == 255) s_total = s_cnt[255];
    __syncthreads();
#pragma unroll
    for (int i = 0; i < 8; ++i) {
        if (loc[i] >= 0) {
            if (off < 256) s_list[off] = loc[i];
            off++;
        }
    }
    __syncthreads();
    int vcount = min(s_total, 256);

    // pad mask
    int* pw = padm + w * SDIM;
    for (int s = tid; s < SDIM; s += 256) {
        bool valid;
        if (s < 256) {
            int iy = y0 + (s >> 4), ix = x0 + (s & 15);
            valid = (iy < PDIM) && (ix < PDIM);
        } else {
            valid = (s - 256) < vcount;
        }
        pw[s] = valid ? 0 : 1;
    }

    // token values
    float* tw = tokens + (size_t)w * SDIM * CDIM;
    for (int i = tid; i < SDIM * CDIM; i += 256) {
        int s = i >> 9, c = i & 511;
        float val = 0.0f;
        if (s < 256) {
            int iy = y0 + (s >> 4), ix = x0 + (s & 15);
            if (iy < PDIM && ix < PDIM)
                val = patch[(size_t)(iy * PDIM + ix) * CDIM + c];
        } else {
            int j = s - 256;
            if (j < vcount) val = vox[(size_t)s_list[j] * CDIM + c];
        }
        tw[i] = val;
    }
}

// ---------------------------------------------------------------------------
// k_layernorm: one wave (32 lanes) per 512-wide row
// ---------------------------------------------------------------------------
__global__ void k_layernorm(const float* __restrict__ x,
                            const float* __restrict__ g,
                            const float* __restrict__ b,
                            float* __restrict__ y, int rows) {
    int wave = threadIdx.x >> 5, lane = threadIdx.x & 31;
    int row = blockIdx.x * 8 + wave;
    if (row >= rows) return;
    const float* xr = x + (size_t)row * CDIM;
    float s = 0.0f, s2 = 0.0f;
    for (int c = lane; c < CDIM; c += 32) {
        float v = xr[c];
        s += v;
        s2 += v * v;
    }
#pragma unroll
    for (int m = 16; m; m >>= 1) {
        s += __shfl_xor(s, m, 32);
        s2 += __shfl_xor(s2, m, 32);
    }
    float mean = s * (1.0f / CDIM);
    float var = s2 * (1.0f / CDIM) - mean * mean;
    float inv = 1.0f / sqrtf(var + 1e-5f);
    float* yr = y + (size_t)row * CDIM;
    for (int c = lane; c < CDIM; c += 32)
        yr[c] = (xr[c] - mean) * inv * g[c] + b[c];
}

// ---------------------------------------------------------------------------
// k_gemm: Y = X(M x 512) @ W(512 x 512) + bias [+ R], WMMA split-f16
// grid: x = 8 col blocks, y = M/64 row blocks; 256 threads (8 waves)
// X tile streamed with async global->LDS (B128); W tile staged pre-split
// hi/lo f16, N-major so B fragments are pure 32-bit DS loads.
// ---------------------------------------------------------------------------
__global__ void k_gemm(const float* __restrict__ X, const float* __restrict__ W,
                       const float* __restrict__ bias,
                       const float* __restrict__ R, float* __restrict__ Y) {
    __shared__ float xs[64][36];          // rows 144B (16B-aligned for B128)
    __shared__ _Float16 wth[64][34];      // [col][k] hi plane
    __shared__ _Float16 wtl[64][34];      // [col][k] lo plane
    int tid = threadIdx.x;
    int wave = tid >> 5, lane = tid & 31, half = lane >> 4, l15 = lane & 15;
    int wrow = wave >> 1, wcol = wave & 1;
    int rowBase = blockIdx.y * 64;
    int colBlock = blockIdx.x * 64;
    int colLocal = wcol * 32;
    unsigned xs_base = lds_offset_of(&xs[0][0]);

    v8f acc0 = {}, acc1 = {};
    for (int kb = 0; kb < 16; ++kb) {
        // X tile 64x32 via async copies: 512 16B chunks, 2 per thread
#pragma unroll
        for (int i = 0; i < 2; ++i) {
            int ch = tid + i * 256;
            int r = ch >> 3, c4 = (ch & 7) * 4;
            async_load_b128(X + (size_t)(rowBase + r) * CDIM + kb * 32 + c4,
                            xs_base + (unsigned)(r * 36 + c4) * 4u);
        }
        // W tile 32x64: coalesced read, split once, store transposed
#pragma unroll
        for (int i = 0; i < 8; ++i) {
            int idx = tid + i * 256;
            int r = idx >> 6, c = idx & 63;  // r = k, c = col
            float v = W[(size_t)(kb * 32 + r) * CDIM + colBlock + c];
            _Float16 h = (_Float16)v;
            wth[c][r] = h;
            wtl[c][r] = (_Float16)(v - (float)h);
        }
        async_wait0();
        __syncthreads();
        Frag a = load_fragA(&xs[wrow * 16][0], 36, l15, 0, half);
        Frag b0 = load_fragB16(&wth[0][0], &wtl[0][0], 34, colLocal + l15, 0, half);
        Frag b1 = load_fragB16(&wth[0][0], &wtl[0][0], 34, colLocal + 16 + l15, 0, half);
        acc0 = mma3(a, b0, acc0);
        acc1 = mma3(a, b1, acc1);
        __syncthreads();
    }
    int orow = rowBase + wrow * 16;
    int c0 = colBlock + colLocal + l15;
    int c1 = c0 + 16;
#pragma unroll
    for (int r = 0; r < 8; ++r) {
        int rr = orow + r + 8 * half;
        size_t rb = (size_t)rr * CDIM;
        float v0 = acc0[r] + bias[c0];
        float v1 = acc1[r] + bias[c1];
        if (R) { v0 += R[rb + c0]; v1 += R[rb + c1]; }
        Y[rb + c0] = v0;
        Y[rb + c1] = v1;
    }
}

// ---------------------------------------------------------------------------
// k_attn: per (window, head, 16-row q tile). 128 threads = 4 waves.
// K/V tiles streamed with async global->LDS B128 copies.
// ---------------------------------------------------------------------------
__global__ void k_attn(const float* __restrict__ q, const float* __restrict__ k,
                       const float* __restrict__ v, const int* __restrict__ padm,
                       float* __restrict__ o) {
    __shared__ float qs[16][65];
    __shared__ float ss[16][513];
    __shared__ float kts[64][68];  // rows 272B (16B-aligned for B128)
    int tid = threadIdx.x;
    int wave = tid >> 5, lane = tid & 31, half = lane >> 4, l15 = lane & 15;
    int w = blockIdx.z, head = blockIdx.y, qb = blockIdx.x * 16;
    size_t baseRow = (size_t)w * SDIM;
    int hcol = head * 64;
    const float scale = 0.125f;  // (C/HEADS)^-0.5 = 64^-0.5
    unsigned kts_base = lds_offset_of(&kts[0][0]);

    for (int i = tid; i < 16 * 64; i += 128) {
        int r = i >> 6, c = i & 63;
        qs[r][c] = q[(baseRow + qb + r) * CDIM + hcol + c] * scale;
    }
    const int* pw = padm + w * SDIM;
    __syncthreads();

    // scores = Q @ K^T  (masked)
    for (int kt = 0; kt < 8; ++kt) {
#pragma unroll
        for (int i = 0; i < 8; ++i) {  // 64x64 tile = 1024 16B chunks
            int ch = tid + i * 128;
            int r = ch >> 4, c4 = (ch & 15) * 4;
            async_load_b128(k + (baseRow + kt * 64 + r) * CDIM + hcol + c4,
                            kts_base + (unsigned)(r * 68 + c4) * 4u);
        }
        async_wait0();
        __syncthreads();
        v8f acc = {};
#pragma unroll
        for (int dk = 0; dk < 2; ++dk) {
            Frag a = load_fragA(&qs[0][0], 65, l15, dk * 32, half);
            Frag b = load_fragBT(&kts[wave * 16][0], 68, l15, dk * 32, half);
            acc = mma3(a, b, acc);
        }
        int col = kt * 64 + wave * 16 + l15;
        bool masked = pw[col] != 0;
#pragma unroll
        for (int r = 0; r < 8; ++r)
            ss[r + 8 * half][col] = masked ? -1e30f : acc[r];
        __syncthreads();
    }

    // softmax (4 waves x 4 rows)
#pragma unroll
    for (int rr = 0; rr < 4; ++rr) {
        int row = wave * 4 + rr;
        float m = -3.0e38f;
        for (int c = lane; c < SDIM; c += 32) m = fmaxf(m, ss[row][c]);
#pragma unroll
        for (int sh = 16; sh; sh >>= 1) m = fmaxf(m, __shfl_xor(m, sh, 32));
        float sum = 0.0f;
        for (int c = lane; c < SDIM; c += 32) {
            float e = expf(ss[row][c] - m);
            ss[row][c] = e;
            sum += e;
        }
#pragma unroll
        for (int sh = 16; sh; sh >>= 1) sum += __shfl_xor(sum, sh, 32);
        float inv = 1.0f / sum;
        for (int c = lane; c < SDIM; c += 32) ss[row][c] *= inv;
    }
    __syncthreads();

    // out = attn @ V ; each wave owns 16 head-dim cols
    v8f oacc = {};
    for (int kt = 0; kt < 8; ++kt) {
#pragma unroll
        for (int i = 0; i < 8; ++i) {
            int ch = tid + i * 128;
            int r = ch >> 4, c4 = (ch & 15) * 4;
            async_load_b128(v + (baseRow + kt * 64 + r) * CDIM + hcol + c4,
                            kts_base + (unsigned)(r * 68 + c4) * 4u);
        }
        async_wait0();
        __syncthreads();
#pragma unroll
        for (int dk = 0; dk < 2; ++dk) {
            Frag a = load_fragA(&ss[0][0], 513, l15, kt * 64 + dk * 32, half);
            Frag b = load_fragB(&kts[0][0], 68, wave * 16 + l15, dk * 32, half);
            oacc = mma3(a, b, oacc);
        }
        __syncthreads();
    }
#pragma unroll
    for (int r = 0; r < 8; ++r)
        o[(baseRow + qb + r + 8 * half) * CDIM + hcol + wave * 16 + l15] = oacc[r];
}

// ---------------------------------------------------------------------------
// k_mlp: x += gelu(lnx @ W1 + b1) @ W2 + b2, hidden chunked at 256.
// 2048 blocks (16 rows each), 256 threads = 8 waves; wave owns 64 out cols.
// ---------------------------------------------------------------------------
__global__ void k_mlp(const float* __restrict__ lnx, const float* __restrict__ w1,
                      const float* __restrict__ b1, const float* __restrict__ w2,
                      const float* __restrict__ b2, float* __restrict__ x) {
    __shared__ float xs[16][513];
    __shared__ float ts[16][257];
    int tid = threadIdx.x;
    int wave = tid >> 5, lane = tid & 31, half = lane >> 4, l15 = lane & 15;
    int rowBase = blockIdx.x * 16;
    int myCol = wave * 64;

    for (int i = tid; i < 16 * CDIM; i += 256) {
        int r = i >> 9, c = i & 511;
        xs[r][c] = lnx[(size_t)(rowBase + r) * CDIM + c];
    }
    __syncthreads();

    v8f out0 = {}, out1 = {}, out2 = {}, out3 = {};
    for (int ch = 0; ch < 8; ++ch) {
        int hb = ch * 256;
        // phase A: t = xs @ W1[:, hb + wave*32 .. +32]
        v8f t0 = {}, t1 = {};
        for (int kb = 0; kb < 16; ++kb) {
            Frag a = load_fragA(&xs[0][0], 513, l15, kb * 32, half);
            Frag bb0 = load_fragB(w1, 2048, hb + wave * 32 + l15, kb * 32, half);
            Frag bb1 = load_fragB(w1, 2048, hb + wave * 32 + 16 + l15, kb * 32, half);
            t0 = mma3(a, bb0, t0);
            t1 = mma3(a, bb1, t1);
        }
        __syncthreads();  // previous chunk's phase B done reading ts
        {
            int row = 8 * half;
            int c0 = wave * 32 + l15, c1 = c0 + 16;
#pragma unroll
            for (int r = 0; r < 8; ++r) {
                ts[row + r][c0] = gelu_exact(t0[r] + b1[hb + c0]);
                ts[row + r][c1] = gelu_exact(t1[r] + b1[hb + c1]);
            }
        }
        __syncthreads();
        // phase B: out += gelu_t @ W2[hb..hb+256, myCol..myCol+64]
        for (int kb = 0; kb < 8; ++kb) {
            Frag a = load_fragA(&ts[0][0], 257, l15, kb * 32, half);
            Frag c0 = load_fragB(w2, CDIM, myCol + l15, hb + kb * 32, half);
            Frag c1 = load_fragB(w2, CDIM, myCol + 16 + l15, hb + kb * 32, half);
            Frag c2 = load_fragB(w2, CDIM, myCol + 32 + l15, hb + kb * 32, half);
            Frag c3 = load_fragB(w2, CDIM, myCol + 48 + l15, hb + kb * 32, half);
            out0 = mma3(a, c0, out0);
            out1 = mma3(a, c1, out1);
            out2 = mma3(a, c2, out2);
            out3 = mma3(a, c3, out3);
        }
    }
#pragma unroll
    for (int r = 0; r < 8; ++r) {
        int rr = rowBase + r + 8 * half;
        size_t rb = (size_t)rr * CDIM;
        int c0 = myCol + l15;
        x[rb + c0]      += out0[r] + b2[c0];
        x[rb + c0 + 16] += out1[r] + b2[c0 + 16];
        x[rb + c0 + 32] += out2[r] + b2[c0 + 32];
        x[rb + c0 + 48] += out3[r] + b2[c0 + 48];
    }
}

// ---------------------------------------------------------------------------
// k_scatter: accumulate valid patch-row outputs into grid + counts
// ---------------------------------------------------------------------------
__global__ void k_scatter(const float* __restrict__ x, float* __restrict__ out,
                          float* __restrict__ counts) {
    int w = blockIdx.x, tid = threadIdx.x;
    int y0 = (w >> 3) * 8, x0 = (w & 7) * 8;
    for (int s = tid; s < 256; s += 256) {
        int iy = y0 + (s >> 4), ix = x0 + (s & 15);
        if (iy < PDIM && ix < PDIM) atomicAdd(&counts[iy * PDIM + ix], 1.0f);
    }
    for (int i = tid; i < 256 * CDIM; i += 256) {
        int s = i >> 9, c = i & 511;
        int iy = y0 + (s >> 4), ix = x0 + (s & 15);
        if (iy < PDIM && ix < PDIM)
            atomicAdd(&out[(size_t)(iy * PDIM + ix) * CDIM + c],
                      x[(size_t)w * SDIM * CDIM + (size_t)s * CDIM + c]);
    }
}

__global__ void k_div(float* __restrict__ out, const float* __restrict__ counts) {
    int i = blockIdx.x * 256 + threadIdx.x;
    if (i < PDIM * PDIM * CDIM) out[i] /= fmaxf(counts[i >> 9], 1.0f);
}

// ---------------------------------------------------------------------------
// host entry
// ---------------------------------------------------------------------------
extern "C" void kernel_launch(void* const* d_in, const int* in_sizes, int n_in,
                              void* d_out, int out_size, void* d_ws, size_t ws_size,
                              hipStream_t stream) {
    (void)in_sizes; (void)n_in; (void)out_size; (void)ws_size;
    const float* patch = (const float*)d_in[0];
    const float* vox   = (const float*)d_in[1];
    const float* ln1_g = (const float*)d_in[2];
    const float* ln1_b = (const float*)d_in[3];
    const float* wq    = (const float*)d_in[4];
    const float* bq    = (const float*)d_in[5];
    const float* wk    = (const float*)d_in[6];
    const float* bk    = (const float*)d_in[7];
    const float* wv    = (const float*)d_in[8];
    const float* bv    = (const float*)d_in[9];
    const float* wo    = (const float*)d_in[10];
    const float* bo    = (const float*)d_in[11];
    const float* ln2_g = (const float*)d_in[12];
    const float* ln2_b = (const float*)d_in[13];
    const float* w1    = (const float*)d_in[14];
    const float* b1    = (const float*)d_in[15];
    const float* w2    = (const float*)d_in[16];
    const float* b2    = (const float*)d_in[17];
    const int*   vpy   = (const int*)d_in[18];
    const int*   vpx   = (const int*)d_in[19];

    const size_t BUFE = (size_t)NROWS * CDIM;  // 16,777,216 floats
    float* buf0 = (float*)d_ws;        // tokens -> x
    float* buf1 = buf0 + BUFE;         // h (LN1) -> attn out o
    float* buf2 = buf1 + BUFE;         // q -> lnx
    float* buf3 = buf2 + BUFE;         // k
    float* buf4 = buf3 + BUFE;         // v
    int*   padm = (int*)(buf4 + BUFE); // [NWIN][SDIM]
    float* counts = (float*)(padm + NWIN * SDIM);
    float* out = (float*)d_out;

    k_zero<<<(PDIM * PDIM * CDIM + 255) / 256, 256, 0, stream>>>(out, counts);
    k_gather<<<NWIN, 256, 0, stream>>>(patch, vox, vpy, vpx, buf0, padm);
    k_layernorm<<<NROWS / 8, 256, 0, stream>>>(buf0, ln1_g, ln1_b, buf1, NROWS);

    dim3 gg(CDIM / 64, NROWS / 64);
    k_gemm<<<gg, 256, 0, stream>>>(buf1, wq, bq, nullptr, buf2);
    k_gemm<<<gg, 256, 0, stream>>>(buf1, wk, bk, nullptr, buf3);
    k_gemm<<<gg, 256, 0, stream>>>(buf1, wv, bv, nullptr, buf4);

    k_attn<<<dim3(SDIM / 16, 8, NWIN), 128, 0, stream>>>(buf2, buf3, buf4, padm, buf1);

    k_gemm<<<gg, 256, 0, stream>>>(buf1, wo, bo, buf0, buf0);       // x = tokens + o@wo+bo
    k_layernorm<<<NROWS / 8, 256, 0, stream>>>(buf0, ln2_g, ln2_b, buf2, NROWS);
    k_mlp<<<NROWS / 16, 256, 0, stream>>>(buf2, w1, b1, w2, b2, buf0);

    k_scatter<<<NWIN, 256, 0, stream>>>(buf0, out, counts);
    k_div<<<(PDIM * PDIM * CDIM + 255) / 256, 256, 0, stream>>>(out, counts);
}
